// GCN_55800215109852
// MI455X (gfx1250) — matrix operand order repaired and checked by reading, not compiled, not run on hardware
//
#include <hip/hip_runtime.h>
#include <hip/hip_bf16.h>

// ---------------- problem constants (match reference) ----------------
#define N_NODES   65536
#define N_EDGES   1048576
#define N_BATCH   1024
#define FEAT      128
#define HID       128
#define OUT_DIM   64
#define EDIM      16
#define NUM_LAYERS 6
#define NUM_T     6
#define NEG       0.01f
#define K1PAD     160   // gate_lin1 K = HID+EDIM = 144, padded to 32-multiple

typedef _Float16 h16_t;
typedef __attribute__((ext_vector_type(16))) _Float16 v16h;
typedef __attribute__((ext_vector_type(8)))  float    v8f;

// ---------------- WMMA fragment loaders (CDNA5 16x16x32 f16 layouts) ----------------
// A 16x32 f16: lane L -> row M = L%16; h = L>>4;
//   VGPR v (v<4): K = h*8 + 2v,+1 ; (v>=4): K = 16 + h*8 + 2(v-4),+1
__device__ __forceinline__ v16h frag_a(const h16_t* __restrict__ row, int h, int k0) {
  v16h f;
#pragma unroll
  for (int v = 0; v < 8; ++v) {
    int K = k0 + ((v < 4) ? (h * 8 + 2 * v) : (16 + h * 8 + 2 * (v - 4)));
    f[2 * v]     = row[K];
    f[2 * v + 1] = row[K + 1];
  }
  return f;
}
// B 32x16 f16: lane L -> col N = L%16; h = L>>4; VGPR v (v<4): K = h*8 + 2v,+1 ; (v>=4): K = 16 + h*8 + 2(v-4),+1
// B[k][n] = W[n][k] for row-major W[Nc,K], so lane reads W row (n0 + L%16).
__device__ __forceinline__ v16h frag_b(const h16_t* __restrict__ wrow, int h, int k0) {
  v16h f;
#pragma unroll
  for (int v = 0; v < 8; ++v) {
    int K = k0 + ((v < 4) ? (h * 8 + 2 * v) : (16 + h * 8 + 2 * (v - 4)));
    f[2 * v]     = wrow[K];
    f[2 * v + 1] = wrow[K + 1];
  }
  return f;
}
// concat(x_row[0:128], edge_attr_row[0:16], zeros[16]) element fetch
__device__ __forceinline__ h16_t cat_elem(const h16_t* __restrict__ x,
                                          const h16_t* __restrict__ e, int K) {
  return (K < HID) ? x[K] : ((K < HID + EDIM) ? e[K - HID] : (h16_t)0.f);
}
__device__ __forceinline__ v16h frag_a_cat(const h16_t* __restrict__ xrow,
                                           const h16_t* __restrict__ erow, int h, int k0) {
  v16h f;
#pragma unroll
  for (int v = 0; v < 8; ++v) {
    int K = k0 + ((v < 4) ? (h * 8 + 2 * v) : (16 + h * 8 + 2 * (v - 4)));
    f[2 * v]     = cat_elem(xrow, erow, K);
    f[2 * v + 1] = cat_elem(xrow, erow, K + 1);
  }
  return f;
}

// ---------------- generic WMMA GEMM: out = act(A[M,K] @ W[Nc,K]^T + bias) ----------------
// act: 0=none 1=leaky_relu 2=relu.  outf/outh optional (f32 and/or f16 copies).
__global__ void k_gemm(const h16_t* __restrict__ A, const h16_t* __restrict__ W,
                       const float* __restrict__ bias, float* __restrict__ outf,
                       h16_t* __restrict__ outh, int M, int Nc, int K, int act) {
  int wave = (blockIdx.x * blockDim.x + threadIdx.x) >> 5;
  int lane = threadIdx.x & 31;
  int tilesN = Nc >> 4;
  int tiles  = (M >> 4) * tilesN;
  if (wave >= tiles) return;                       // wave-uniform: EXEC all-1s inside
  int m0 = (wave / tilesN) << 4;
  int n0 = (wave % tilesN) << 4;
  int c = lane & 15, h = lane >> 4;
  const h16_t* arow = A + (size_t)(m0 + c) * K;
  const h16_t* wrow = W + (size_t)(n0 + c) * K;
  v8f acc = {};
  for (int k0 = 0; k0 < K; k0 += 32) {
    v16h a = frag_a(arow, h, k0);
    v16h b = frag_b(wrow, h, k0);
    acc = __builtin_amdgcn_wmma_f32_16x16x32_f16(false, a, false, b, (short)0, acc, false, false);
  }
  float bv = bias ? bias[n0 + c] : 0.f;
#pragma unroll
  for (int v = 0; v < 8; ++v) {
    float val = acc[v] + bv;
    if (act == 1)      val = (val > 0.f) ? val : NEG * val;
    else if (act == 2) val = fmaxf(val, 0.f);
    size_t o = (size_t)(m0 + v + 8 * h) * Nc + (n0 + c);
    if (outf) outf[o] = val;
    if (outh) outh[o] = (h16_t)val;
  }
}

// ---------------- GATEConv edge transform: xj = leaky([x[src]||ea] @ W1pad^T) ----------------
__global__ void k_xj(const h16_t* __restrict__ x16, const h16_t* __restrict__ ea16,
                     const int* __restrict__ src, const h16_t* __restrict__ W1,
                     h16_t* __restrict__ xj) {
  int wave = (blockIdx.x * blockDim.x + threadIdx.x) >> 5;
  int lane = threadIdx.x & 31;
  const int tilesN = HID >> 4;
  const int tiles  = (N_EDGES >> 4) * tilesN;
  if (wave >= tiles) return;
  int m0 = (wave / tilesN) << 4;
  int n0 = (wave % tilesN) << 4;
  int c = lane & 15, h = lane >> 4;
  int erow = m0 + c;
  const h16_t* xrow = x16 + (size_t)src[erow] * HID;   // gathered A row
  const h16_t* eat  = ea16 + (size_t)erow * EDIM;
  const h16_t* wrow = W1 + (size_t)(n0 + c) * K1PAD;
  v8f acc = {};
#pragma unroll
  for (int k0 = 0; k0 < K1PAD; k0 += 32) {
    v16h a = frag_a_cat(xrow, eat, h, k0);
    v16h b = frag_b(wrow, h, k0);
    acc = __builtin_amdgcn_wmma_f32_16x16x32_f16(false, a, false, b, (short)0, acc, false, false);
  }
#pragma unroll
  for (int v = 0; v < 8; ++v) {
    float val = acc[v];
    val = (val > 0.f) ? val : NEG * val;
    xj[(size_t)(m0 + v + 8 * h) * HID + (n0 + c)] = (h16_t)val;
  }
}

// ------- fused message GEMM + scatter: h[dst[e]] += alpha[e] * (xj[e] @ W2^T) -------
__global__ void k_msg(const h16_t* __restrict__ xj, const h16_t* __restrict__ W2,
                      const float* __restrict__ alpha, const int* __restrict__ dst,
                      float* __restrict__ hbuf) {
  int wave = (blockIdx.x * blockDim.x + threadIdx.x) >> 5;
  int lane = threadIdx.x & 31;
  const int tilesN = HID >> 4;
  const int tiles  = (N_EDGES >> 4) * tilesN;
  if (wave >= tiles) return;
  int m0 = (wave / tilesN) << 4;
  int n0 = (wave % tilesN) << 4;
  int c = lane & 15, h = lane >> 4;
  const h16_t* arow = xj + (size_t)(m0 + c) * HID;
  const h16_t* wrow = W2 + (size_t)(n0 + c) * HID;
  v8f acc = {};
#pragma unroll
  for (int k0 = 0; k0 < HID; k0 += 32) {
    v16h a = frag_a(arow, h, k0);
    v16h b = frag_b(wrow, h, k0);
    acc = __builtin_amdgcn_wmma_f32_16x16x32_f16(false, a, false, b, (short)0, acc, false, false);
  }
#pragma unroll
  for (int v = 0; v < 8; ++v) {
    int e = m0 + v + 8 * h;
    atomicAdd(&hbuf[(size_t)dst[e] * HID + (n0 + c)], acc[v] * alpha[e]);
  }
}

// ---------------- small helper kernels ----------------
__global__ void k_pack(const float* __restrict__ s, h16_t* __restrict__ d,
                       int rows, int K, int Kpad) {
  int t = blockIdx.x * blockDim.x + threadIdx.x;
  if (t >= rows * Kpad) return;
  int r = t / Kpad, k = t - r * Kpad;
  d[t] = (k < K) ? (h16_t)s[(size_t)r * K + k] : (h16_t)0.f;
}
__global__ void k_rowdot(const float* __restrict__ X, const float* __restrict__ v,
                         float* __restrict__ out, int M) {
  int r = blockIdx.x * blockDim.x + threadIdx.x;
  if (r >= M) return;
  const float* p = X + (size_t)r * HID;
  float s = 0.f;
#pragma unroll 8
  for (int k = 0; k < HID; ++k) s += p[k] * v[k];
  out[r] = s;
}
__global__ void k_alpha_gate(const h16_t* __restrict__ xj, const float* __restrict__ attl,
                             const float* __restrict__ xr, const int* __restrict__ dst,
                             float* __restrict__ alpha) {
  int e = blockIdx.x * blockDim.x + threadIdx.x;
  if (e >= N_EDGES) return;
  const h16_t* p = xj + (size_t)e * HID;
  float s = 0.f;
#pragma unroll 8
  for (int k = 0; k < HID; ++k) s += (float)p[k] * attl[k];
  s += xr[dst[e]];
  alpha[e] = (s > 0.f) ? s : NEG * s;
}
__global__ void k_alpha_sum(const float* __restrict__ a, const int* __restrict__ ia,
                            const float* __restrict__ b, const int* __restrict__ ib,
                            float* __restrict__ alpha, int M) {
  int t = blockIdx.x * blockDim.x + threadIdx.x;
  if (t >= M) return;
  float s = a[ia ? ia[t] : t] + b[ib[t]];
  alpha[t] = (s > 0.f) ? s : NEG * s;
}

// float-ordered unsigned encoding for atomicMax-based segment max
__device__ __forceinline__ unsigned fenc(float f) {
  int i = __float_as_int(f);
  return (i >= 0) ? ((unsigned)i ^ 0x80000000u) : ~(unsigned)i;
}
__device__ __forceinline__ float fdec(unsigned k) {
  int i = (k & 0x80000000u) ? (int)(k ^ 0x80000000u) : ~(int)k;
  return __int_as_float(i);
}
__global__ void k_fillu(unsigned* p, unsigned v, int n) {
  int t = blockIdx.x * blockDim.x + threadIdx.x;
  if (t < n) p[t] = v;
}
__global__ void k_segmax(const float* __restrict__ alpha, const int* __restrict__ seg,
                         unsigned* __restrict__ smax, int M) {
  int t = blockIdx.x * blockDim.x + threadIdx.x;
  if (t >= M) return;
  atomicMax(&smax[seg[t]], fenc(alpha[t]));
}
__global__ void k_segexp(float* __restrict__ alpha, const int* __restrict__ seg,
                         const unsigned* __restrict__ smax, float* __restrict__ ssum, int M) {
  int t = blockIdx.x * blockDim.x + threadIdx.x;
  if (t >= M) return;
  float m = fdec(smax[seg[t]]);
  if (!(m > -3.0e38f && m < 3.0e38f)) m = 0.f;   // empty segment -> 0 (matches ref)
  float e = __expf(alpha[t] - m);
  alpha[t] = e;
  atomicAdd(&ssum[seg[t]], e);
}
__global__ void k_segnorm(float* __restrict__ alpha, const int* __restrict__ seg,
                          const float* __restrict__ ssum, int M) {
  int t = blockIdx.x * blockDim.x + threadIdx.x;
  if (t >= M) return;
  alpha[t] = alpha[t] / (ssum[seg[t]] + 1e-16f);
}
// out[dst[r]] += alpha[r] * feat16[src[r]]  (src==nullptr -> identity; alpha==nullptr -> 1)
__global__ void k_scatter(const h16_t* __restrict__ feat, const float* __restrict__ alpha,
                          const int* __restrict__ sidx, const int* __restrict__ didx,
                          float* __restrict__ out, int M) {
  int t = blockIdx.x * blockDim.x + threadIdx.x;
  int r = t >> 3;
  if (r >= M) return;
  int f0 = (t & 7) << 4;
  int sr = sidx ? sidx[r] : r;
  float a = alpha ? alpha[r] : 1.f;
  const h16_t* p = feat + (size_t)sr * HID + f0;
  float* q = out + (size_t)didx[r] * HID + f0;
#pragma unroll
  for (int i = 0; i < 16; ++i) atomicAdd(q + i, a * (float)p[i]);
}
__global__ void k_elu_bias(const float* __restrict__ in, const float* __restrict__ bias,
                           float* __restrict__ of, h16_t* __restrict__ oh, int n) {
  int t = blockIdx.x * blockDim.x + threadIdx.x;
  if (t >= n) return;
  float v = in[t] + bias[t & (HID - 1)];
  v = (v > 0.f) ? v : (__expf(v) - 1.f);
  of[t] = v;
  oh[t] = (h16_t)v;
}
__global__ void k_relu_ew(float* __restrict__ io, h16_t* __restrict__ oh, int n) {
  int t = blockIdx.x * blockDim.x + threadIdx.x;
  if (t >= n) return;
  float v = fmaxf(io[t], 0.f);
  io[t] = v;
  oh[t] = (h16_t)v;
}
__device__ __forceinline__ float sigm(float x) { return 1.f / (1.f + __expf(-x)); }
// GRUCell(inp=h, hid=x) followed by relu; writes f32 + f16 copies (may alias hidden)
__global__ void k_gru(const float* __restrict__ gi, const float* __restrict__ gh,
                      const float* __restrict__ hidden, float* __restrict__ xo,
                      h16_t* __restrict__ xo16, int M) {
  int t = blockIdx.x * blockDim.x + threadIdx.x;
  if (t >= M * HID) return;
  int r = t >> 7, j = t & (HID - 1);
  const float* gir = gi + (size_t)r * 3 * HID;
  const float* ghr = gh + (size_t)r * 3 * HID;
  float rg = sigm(gir[j] + ghr[j]);
  float z  = sigm(gir[HID + j] + ghr[HID + j]);
  float n  = tanhf(gir[2 * HID + j] + rg * ghr[2 * HID + j]);
  float hv = hidden[t];
  float v  = fmaxf((1.f - z) * n + z * hv, 0.f);
  xo[t]  = v;
  xo16[t] = (h16_t)v;
}

// ---------------- host-side helpers ----------------
static inline char* bump(char*& p, size_t bytes) {
  char* r = p;
  p += (bytes + 255) & ~(size_t)255;
  return r;
}
static inline void gemm(hipStream_t s, const h16_t* A, const h16_t* W, const float* b,
                        float* of, h16_t* oh, int M, int Nc, int K, int act) {
  int tiles = (M >> 4) * (Nc >> 4);
  k_gemm<<<(tiles + 3) >> 2, 128, 0, s>>>(A, W, b, of, oh, M, Nc, K, act);
}
static inline void segsoftmax(hipStream_t s, float* alpha, const int* seg, int M, int nseg,
                              unsigned* smax, float* ssum) {
  k_fillu<<<(nseg + 255) / 256, 256, 0, s>>>(smax, 0x007FFFFFu, nseg);  // enc(-inf)
  hipMemsetAsync(ssum, 0, (size_t)nseg * sizeof(float), s);
  k_segmax<<<(M + 255) / 256, 256, 0, s>>>(alpha, seg, smax, M);
  k_segexp<<<(M + 255) / 256, 256, 0, s>>>(alpha, seg, smax, ssum, M);
  k_segnorm<<<(M + 255) / 256, 256, 0, s>>>(alpha, seg, ssum, M);
}
static inline void gru_step(hipStream_t s, const h16_t* in16, const h16_t* hid16,
                            const float* hidf, const h16_t* Wih, const h16_t* Whh,
                            const float* bih, const float* bhh, float* gi, float* gh,
                            float* xo, h16_t* xo16, int M) {
  gemm(s, in16,  Wih, bih, gi, nullptr, M, 3 * HID, HID, 0);
  gemm(s, hid16, Whh, bhh, gh, nullptr, M, 3 * HID, HID, 0);
  k_gru<<<((M * HID) + 255) / 256, 256, 0, s>>>(gi, gh, hidf, xo, xo16, M);
}

extern "C" void kernel_launch(void* const* d_in, const int* in_sizes, int n_in,
                              void* d_out, int out_size, void* d_ws, size_t ws_size,
                              hipStream_t stream) {
  (void)in_sizes; (void)n_in; (void)out_size; (void)ws_size;
  // ---- inputs (setup_inputs order) ----
  const float* x_in      = (const float*)d_in[0];
  const float* edge_attr = (const float*)d_in[1];
  const float* lin1_W    = (const float*)d_in[2];
  const float* lin1_b    = (const float*)d_in[3];
  const float* gate_att_l= (const float*)d_in[4];
  const float* gate_att_r= (const float*)d_in[5];
  const float* gate_lin1_W=(const float*)d_in[6];
  const float* gate_lin2_W=(const float*)d_in[7];
  const float* gate_bias = (const float*)d_in[8];
  const float* gat_W     = (const float*)d_in[9];
  const float* gat_att_src=(const float*)d_in[10];
  const float* gat_att_dst=(const float*)d_in[11];
  const float* gat_bias  = (const float*)d_in[12];
  const float* gru_Wih   = (const float*)d_in[13];
  const float* gru_Whh   = (const float*)d_in[14];
  const float* gru_bih   = (const float*)d_in[15];
  const float* gru_bhh   = (const float*)d_in[16];
  const float* mol_W     = (const float*)d_in[17];
  const float* mol_att_src=(const float*)d_in[18];
  const float* mol_att_dst=(const float*)d_in[19];
  const float* mol_bias  = (const float*)d_in[20];
  const float* mol_gru_Wih=(const float*)d_in[21];
  const float* mol_gru_Whh=(const float*)d_in[22];
  const float* mol_gru_bih=(const float*)d_in[23];
  const float* mol_gru_bhh=(const float*)d_in[24];
  const float* cls_W     = (const float*)d_in[25];
  const float* cls_b     = (const float*)d_in[26];
  const float* lin2_W    = (const float*)d_in[27];
  const float* lin2_b    = (const float*)d_in[28];
  const int*   src       = (const int*)d_in[29];            // edge_index[0]
  const int*   dst       = src + N_EDGES;                   // edge_index[1]
  const int*   batch     = (const int*)d_in[30];
  float* out1 = (float*)d_out;                              // [N, OUT]
  float* out2 = out1 + (size_t)N_NODES * OUT_DIM;           // [B, OUT]

  // ---- workspace carve-out ----
  char* p = (char*)d_ws;
  h16_t* w_lin1  = (h16_t*)bump(p, (size_t)HID * FEAT * 2);
  h16_t* w_gate1 = (h16_t*)bump(p, (size_t)HID * K1PAD * 2);
  h16_t* w_gate2 = (h16_t*)bump(p, (size_t)HID * HID * 2);
  h16_t* w_gat   = (h16_t*)bump(p, (size_t)(NUM_LAYERS - 1) * HID * HID * 2);
  h16_t* w_gwih  = (h16_t*)bump(p, (size_t)NUM_LAYERS * 3 * HID * HID * 2);
  h16_t* w_gwhh  = (h16_t*)bump(p, (size_t)NUM_LAYERS * 3 * HID * HID * 2);
  h16_t* w_mol   = (h16_t*)bump(p, (size_t)HID * HID * 2);
  h16_t* w_mwih  = (h16_t*)bump(p, (size_t)3 * HID * HID * 2);
  h16_t* w_mwhh  = (h16_t*)bump(p, (size_t)3 * HID * HID * 2);
  h16_t* w_cls   = (h16_t*)bump(p, (size_t)OUT_DIM * HID * 2);
  h16_t* w_lin2  = (h16_t*)bump(p, (size_t)OUT_DIM * HID * 2);
  h16_t* xin16   = (h16_t*)bump(p, (size_t)N_NODES * FEAT * 2);
  h16_t* x16     = (h16_t*)bump(p, (size_t)N_NODES * HID * 2);
  float* xf      = (float*)bump(p, (size_t)N_NODES * HID * 4);
  float* hf      = (float*)bump(p, (size_t)N_NODES * HID * 4);
  h16_t* h16b    = (h16_t*)bump(p, (size_t)N_NODES * HID * 2);
  float* gi      = (float*)bump(p, (size_t)N_NODES * 3 * HID * 4);
  float* gh      = (float*)bump(p, (size_t)N_NODES * 3 * HID * 4);
  h16_t* ea16    = (h16_t*)bump(p, (size_t)N_EDGES * EDIM * 2);
  h16_t* xj16    = (h16_t*)bump(p, (size_t)N_EDGES * HID * 2);
  float* xlf     = (float*)bump(p, (size_t)N_NODES * HID * 4);
  h16_t* xl16    = (h16_t*)bump(p, (size_t)N_NODES * HID * 2);
  float* alpha   = (float*)bump(p, (size_t)N_EDGES * 4);
  float* nsc0    = (float*)bump(p, (size_t)N_NODES * 4);     // xr / asrc
  float* nsc1    = (float*)bump(p, (size_t)N_NODES * 4);     // adst
  unsigned* smaxN= (unsigned*)bump(p, (size_t)N_NODES * 4);
  float* ssumN   = (float*)bump(p, (size_t)N_NODES * 4);
  float* outBf   = (float*)bump(p, (size_t)N_BATCH * HID * 4);
  h16_t* outB16  = (h16_t*)bump(p, (size_t)N_BATCH * HID * 2);
  float* hBf     = (float*)bump(p, (size_t)N_BATCH * HID * 4);
  h16_t* hB16    = (h16_t*)bump(p, (size_t)N_BATCH * HID * 2);
  float* giB     = (float*)bump(p, (size_t)N_BATCH * 3 * HID * 4);
  float* ghB     = (float*)bump(p, (size_t)N_BATCH * 3 * HID * 4);
  float* xsf     = (float*)bump(p, (size_t)N_NODES * HID * 4);
  h16_t* xs16    = (h16_t*)bump(p, (size_t)N_NODES * HID * 2);
  float* asrcN   = (float*)bump(p, (size_t)N_NODES * 4);
  float* bdst    = (float*)bump(p, (size_t)N_BATCH * 4);
  unsigned* smaxB= (unsigned*)bump(p, (size_t)N_BATCH * 4);
  float* ssumB   = (float*)bump(p, (size_t)N_BATCH * 4);
  float* odf     = (float*)bump(p, (size_t)N_BATCH * HID * 4);

  auto pack = [&](const float* s_, h16_t* d_, int rows, int K, int Kpad) {
    int n = rows * Kpad;
    k_pack<<<(n + 255) / 256, 256, 0, stream>>>(s_, d_, rows, K, Kpad);
  };
  // ---- pack all weights / fp32 activations to fp16 ----
  pack(lin1_W,       w_lin1,  HID, FEAT, FEAT);
  pack(gate_lin1_W,  w_gate1, HID, HID + EDIM, K1PAD);  // zero-pad K 144 -> 160
  pack(gate_lin2_W,  w_gate2, HID, HID, HID);
  pack(gat_W,        w_gat,   (NUM_LAYERS - 1) * HID, HID, HID);
  pack(gru_Wih,      w_gwih,  NUM_LAYERS * 3 * HID, HID, HID);
  pack(gru_Whh,      w_gwhh,  NUM_LAYERS * 3 * HID, HID, HID);
  pack(mol_W,        w_mol,   HID, HID, HID);
  pack(mol_gru_Wih,  w_mwih,  3 * HID, HID, HID);
  pack(mol_gru_Whh,  w_mwhh,  3 * HID, HID, HID);
  pack(cls_W,        w_cls,   OUT_DIM, HID, HID);
  pack(lin2_W,       w_lin2,  OUT_DIM, HID, HID);
  pack(x_in,         xin16,   N_NODES, FEAT, FEAT);
  pack(edge_attr,    ea16,    N_EDGES, EDIM, EDIM);

  // ---- x = leaky_relu(x @ lin1_W^T + b) ----
  gemm(stream, xin16, w_lin1, lin1_b, xf, x16, N_NODES, HID, FEAT, 1);

  // ---- layer 0: GATEConv + GRU ----
  {
    int tiles = (N_EDGES >> 4) * (HID >> 4);
    k_xj<<<(tiles + 3) >> 2, 128, 0, stream>>>(x16, ea16, src, w_gate1, xj16);
    k_rowdot<<<(N_NODES + 255) / 256, 256, 0, stream>>>(xf, gate_att_r, nsc0, N_NODES);
    k_alpha_gate<<<(N_EDGES + 255) / 256, 256, 0, stream>>>(xj16, gate_att_l, nsc0, dst, alpha);
    segsoftmax(stream, alpha, dst, N_EDGES, N_NODES, smaxN, ssumN);
    hipMemsetAsync(hf, 0, (size_t)N_NODES * HID * sizeof(float), stream);
    k_msg<<<(tiles + 3) >> 2, 128, 0, stream>>>(xj16, w_gate2, alpha, dst, hf);
    k_elu_bias<<<((N_NODES * HID) + 255) / 256, 256, 0, stream>>>(hf, gate_bias, hf, h16b, N_NODES * HID);
    gru_step(stream, h16b, x16, xf, w_gwih, w_gwhh, gru_bih, gru_bhh, gi, gh, xf, x16, N_NODES);
  }

  // ---- layers 1..5: GATConv + GRU ----
  for (int l = 0; l < NUM_LAYERS - 1; ++l) {
    gemm(stream, x16, w_gat + (size_t)l * HID * HID, nullptr, xlf, xl16, N_NODES, HID, HID, 0);
    k_rowdot<<<(N_NODES + 255) / 256, 256, 0, stream>>>(xlf, gat_att_src + (size_t)l * HID, nsc0, N_NODES);
    k_rowdot<<<(N_NODES + 255) / 256, 256, 0, stream>>>(xlf, gat_att_dst + (size_t)l * HID, nsc1, N_NODES);
    k_alpha_sum<<<(N_EDGES + 255) / 256, 256, 0, stream>>>(nsc0, src, nsc1, dst, alpha, N_EDGES);
    segsoftmax(stream, alpha, dst, N_EDGES, N_NODES, smaxN, ssumN);
    hipMemsetAsync(hf, 0, (size_t)N_NODES * HID * sizeof(float), stream);
    k_scatter<<<((N_EDGES * 8) + 255) / 256, 256, 0, stream>>>(xl16, alpha, src, dst, hf, N_EDGES);
    k_elu_bias<<<((N_NODES * HID) + 255) / 256, 256, 0, stream>>>(hf, gat_bias + (size_t)l * HID, hf, h16b, N_NODES * HID);
    gru_step(stream, h16b, x16, xf,
             w_gwih + (size_t)(l + 1) * 3 * HID * HID, w_gwhh + (size_t)(l + 1) * 3 * HID * HID,
             gru_bih + (size_t)(l + 1) * 3 * HID, gru_bhh + (size_t)(l + 1) * 3 * HID,
             gi, gh, xf, x16, N_NODES);
  }

  // ---- molecule readout ----
  hipMemsetAsync(outBf, 0, (size_t)N_BATCH * HID * sizeof(float), stream);
  k_scatter<<<((N_NODES * 8) + 255) / 256, 256, 0, stream>>>(x16, nullptr, nullptr, batch, outBf, N_NODES);
  k_relu_ew<<<((N_BATCH * HID) + 255) / 256, 256, 0, stream>>>(outBf, outB16, N_BATCH * HID);
  gemm(stream, x16, w_mol, nullptr, xsf, xs16, N_NODES, HID, HID, 0);  // xs (const over t)
  k_rowdot<<<(N_NODES + 255) / 256, 256, 0, stream>>>(xsf, mol_att_src, asrcN, N_NODES);
  for (int t = 0; t < NUM_T; ++t) {
    gemm(stream, outB16, w_mol, nullptr, odf, nullptr, N_BATCH, HID, HID, 0);
    k_rowdot<<<(N_BATCH + 255) / 256, 256, 0, stream>>>(odf, mol_att_dst, bdst, N_BATCH);
    k_alpha_sum<<<(N_NODES + 255) / 256, 256, 0, stream>>>(asrcN, nullptr, bdst, batch, alpha, N_NODES);
    segsoftmax(stream, alpha, batch, N_NODES, N_BATCH, smaxB, ssumB);
    hipMemsetAsync(hBf, 0, (size_t)N_BATCH * HID * sizeof(float), stream);
    k_scatter<<<((N_NODES * 8) + 255) / 256, 256, 0, stream>>>(xs16, alpha, nullptr, batch, hBf, N_NODES);
    k_elu_bias<<<((N_BATCH * HID) + 255) / 256, 256, 0, stream>>>(hBf, mol_bias, hBf, hB16, N_BATCH * HID);
    gru_step(stream, hB16, outB16, outBf, w_mwih, w_mwhh, mol_gru_bih, mol_gru_bhh,
             giB, ghB, outBf, outB16, N_BATCH);
  }

  // ---- heads ----
  gemm(stream, x16,   w_cls,  cls_b,  out1, nullptr, N_NODES, OUT_DIM, HID, 0);
  gemm(stream, outB16, w_lin2, lin2_b, out2, nullptr, N_BATCH, OUT_DIM, HID, 0);
}
